// RWKV7Attention_76682346103518
// MI455X (gfx1250) — matrix operand order repaired and checked by reading, not compiled, not run on hardware
//
#include <hip/hip_runtime.h>

// ---------------- problem constants ----------------
constexpr int Bb  = 4;
constexpr int Tt  = 4096;
constexpr int Dd  = 1024;
constexpr int Hh  = 16;
constexpr int HDd = 64;
constexpr long long Mrows = (long long)Bb * Tt;          // 16384
constexpr long long MD    = Mrows * Dd;                  // 16,777,216

typedef __attribute__((ext_vector_type(8)))  float  v8f;
typedef __attribute__((ext_vector_type(16))) __bf16 v16bf;

union FragU { struct { uint4 lo, hi; } q; v16bf v; };

// ---------------- helpers ----------------
__device__ __forceinline__ __bf16 f32_to_bf16(float f) {
  union { float f; unsigned u; } x; x.f = f;
  unsigned r = x.u + 0x7FFFu + ((x.u >> 16) & 1u);   // round-to-nearest-even
  unsigned short h = (unsigned short)(r >> 16);
  return __builtin_bit_cast(__bf16, h);
}
__device__ __forceinline__ float fast_rcp(float x) {   // v_rcp_f32 (~1 ULP)
  return __builtin_amdgcn_rcpf(x);
}
__device__ __forceinline__ float sigmoidf_(float x) {
  return fast_rcp(1.0f + __expf(-x));
}
__device__ __forceinline__ float tanhf_(float x) {     // 1 - 2/(e^(2x)+1)
  return 1.0f - 2.0f * fast_rcp(__expf(2.0f * x) + 1.0f);
}
__device__ __forceinline__ float wave_sum(float x) {
  #pragma unroll
  for (int m = 16; m > 0; m >>= 1) x += __shfl_xor(x, m, 32);
  return x;
}

// ---------------- f32 -> bf16 convert (4-wide) ----------------
__global__ __launch_bounds__(256) void cvt_bf16x4_kernel(const float* __restrict__ src,
                                                         __bf16* __restrict__ dst, int n4) {
  int i = blockIdx.x * 256 + threadIdx.x;
  if (i >= n4) return;
  float4 f = ((const float4*)src)[i];
  union { __bf16 e[4]; uint2 u; } p;
  p.e[0] = f32_to_bf16(f.x); p.e[1] = f32_to_bf16(f.y);
  p.e[2] = f32_to_bf16(f.z); p.e[3] = f32_to_bf16(f.w);
  ((uint2*)dst)[i] = p.u;
}

// zero-padded convert: dst[dst_rows][dst_cols], src[src_rows][src_cols]
__global__ __launch_bounds__(256) void cvt_pad_kernel(const float* __restrict__ src,
                                                      __bf16* __restrict__ dst,
                                                      int dst_rows, int dst_cols,
                                                      int src_rows, int src_cols) {
  int i = blockIdx.x * 256 + threadIdx.x;
  if (i >= dst_rows * dst_cols) return;
  int r = i / dst_cols, c = i % dst_cols;
  float v = (r < src_rows && c < src_cols) ? src[r * src_cols + c] : 0.0f;
  dst[i] = f32_to_bf16(v);
}

// ---------------- time-shift mix prep (4-wide): xs_i = x + (shift(x)-x)*x_x[i] ----------------
__global__ __launch_bounds__(256) void prep_kernel(const float* __restrict__ hs,
                                                   const float* __restrict__ xx,
                                                   __bf16* __restrict__ xr, __bf16* __restrict__ xw,
                                                   __bf16* __restrict__ xk, __bf16* __restrict__ xv,
                                                   __bf16* __restrict__ xa, __bf16* __restrict__ xg) {
  constexpr int D4 = Dd / 4;
  long long i4 = (long long)blockIdx.x * 256 + threadIdx.x;
  if (i4 >= MD / 4) return;
  int d4 = (int)(i4 % D4);
  long long m = i4 / D4;
  int t = (int)(m % Tt);
  float4 x = ((const float4*)hs)[i4];
  float4 pv = (t > 0) ? ((const float4*)hs)[i4 - D4] : make_float4(0.f, 0.f, 0.f, 0.f);
  float4 dl = make_float4(pv.x - x.x, pv.y - x.y, pv.z - x.z, pv.w - x.w);
  __bf16* outs[6] = {xr, xw, xk, xv, xa, xg};
  #pragma unroll
  for (int s = 0; s < 6; ++s) {
    float4 c = ((const float4*)xx)[s * D4 + d4];
    union { __bf16 e[4]; uint2 u; } p;
    p.e[0] = f32_to_bf16(x.x + dl.x * c.x);
    p.e[1] = f32_to_bf16(x.y + dl.y * c.y);
    p.e[2] = f32_to_bf16(x.z + dl.z * c.z);
    p.e[3] = f32_to_bf16(x.w + dl.w * c.w);
    ((uint2*)outs[s])[i4] = p.u;
  }
}

// ---------------- WMMA bf16 GEMM ----------------
// Requirements (guaranteed by construction): M%128==0, N%BN==0, K%32==0,
// buffers padded so every tile access is in-bounds.
// Block 128xBN, 8 waves as 4x2 grid, each wave owns a 32x(BN/2) register tile
//   BN=128: 8 v_wmma per wave per K-step (A frags reused over 4 B frags)
//   BN= 64: 4 v_wmma per wave per K-step
// mode 0: f32 store        mode 1: bf16(tanh(c))      mode 2: bf16(c)
// mode 3: bf16(sigm(c))    mode 4: f32 -e^-.5*sigm(c+bias)   mode 5: f32 sigm(c+bias)
__device__ __forceinline__ void gemm_epi(int mode, float* Cf, __bf16* Cb,
                                         const float* bias, float c,
                                         int row, int col, int Ndim) {
  long long idx = (long long)row * Ndim + col;
  switch (mode) {
    case 0: Cf[idx] = c; break;
    case 1: Cb[idx] = f32_to_bf16(tanhf_(c)); break;
    case 2: Cb[idx] = f32_to_bf16(c); break;
    case 3: Cb[idx] = f32_to_bf16(sigmoidf_(c)); break;
    case 4: Cf[idx] = -0.6065306597126334f * sigmoidf_(c + bias[col]); break;
    case 5: Cf[idx] = sigmoidf_(c + bias[col]); break;
  }
}

template <int BN>
__global__ __launch_bounds__(256) void gemm_bf16_wmma(
    const __bf16* __restrict__ A, const __bf16* __restrict__ Bm,
    float* __restrict__ Cf, __bf16* __restrict__ Cb,
    const float* __restrict__ bias,
    int Ndim, int Kdim, int mode) {
  constexpr int WTN = BN / 2;       // wave tile N
  constexpr int NT  = WTN / 16;     // B fragments per wave
  // LDS rows padded to 40 elems (80B = 20 banks) -> conflict-free b128 bursts
  __shared__ __attribute__((aligned(16))) __bf16 As[128][40];  // A tile 128x32
  __shared__ __attribute__((aligned(16))) __bf16 Bt[BN][40];   // B tile transposed: [n][k]

  const int bm = blockIdx.y * 128;
  const int bn = blockIdx.x * BN;
  const int tid  = threadIdx.x;
  const int wave = tid >> 5;
  const int lane = tid & 31;
  const int tr = (wave & 3) * 32;       // wave's 32-row slice
  const int tc = (wave >> 2) * WTN;     // wave's WTN-col slice
  const int lane15 = lane & 15;
  const int half   = lane >> 4;

  v8f acc[2][NT];
  #pragma unroll
  for (int i = 0; i < 2; ++i)
    #pragma unroll
    for (int j = 0; j < NT; ++j)
      acc[i][j] = v8f{0.f,0.f,0.f,0.f,0.f,0.f,0.f,0.f};

  // A-fill: 128x32 elems, 16/thread (two b128)
  const int a_row = tid >> 1;                  // 0..127
  const int a_col = (tid & 1) * 16;            // 0 or 16
  // B-fill: 32xBN elems, BN/8 per thread (8 threads per k-row)
  const int b_row = tid >> 3;                  // 0..31  (k within tile)
  const int b_col = (tid & 7) * (BN / 8);      // n within tile

  for (int k0 = 0; k0 < Kdim; k0 += 32) {
    {
      const __bf16* gA = A + (long long)(bm + a_row) * Kdim + k0 + a_col;
      uint4 qa0 = *(const uint4*)gA;
      uint4 qa1 = *(const uint4*)(gA + 8);
      *(uint4*)&As[a_row][a_col]     = qa0;
      *(uint4*)&As[a_row][a_col + 8] = qa1;

      const __bf16* gB = Bm + (long long)(k0 + b_row) * Ndim + bn + b_col;
      #pragma unroll
      for (int u = 0; u < BN / 64; ++u) {
        union { uint4 q; __bf16 e[8]; } tb;
        tb.q = *(const uint4*)(gB + u * 8);
        #pragma unroll
        for (int j = 0; j < 8; ++j) Bt[b_col + u * 8 + j][b_row] = tb.e[j];
      }
    }
    __syncthreads();

    // A frags: lane half selects K {0..7,16..23} / {8..15,24..31}
    const int kb  = half * 8;
    const int kof = half * 16;
    FragU a0, a1;
    a0.q.lo = *(const uint4*)&As[tr + lane15][kb];
    a0.q.hi = *(const uint4*)&As[tr + lane15][16 + kb];
    a1.q.lo = *(const uint4*)&As[tr + 16 + lane15][kb];
    a1.q.hi = *(const uint4*)&As[tr + 16 + lane15][16 + kb];
    #pragma unroll
    for (int j = 0; j < NT; ++j) {
      FragU bf;
      bf.q.lo = *(const uint4*)&Bt[tc + 16 * j + lane15][kb];
      bf.q.hi = *(const uint4*)&Bt[tc + 16 * j + lane15][16 + kb];
      acc[0][j] = __builtin_amdgcn_wmma_f32_16x16x32_bf16(false, a0.v, false, bf.v,
                                                          (short)0, acc[0][j], false, false);
      acc[1][j] = __builtin_amdgcn_wmma_f32_16x16x32_bf16(false, a1.v, false, bf.v,
                                                          (short)0, acc[1][j], false, false);
    }
    __syncthreads();
  }

  // C layout: element i -> row = i + 8*half, col = lane&15
  const int rbase = bm + tr + (half << 3);
  #pragma unroll
  for (int j = 0; j < NT; ++j) {
    const int cb = bn + tc + 16 * j + lane15;
    #pragma unroll
    for (int i = 0; i < 8; ++i) {
      gemm_epi(mode, Cf, Cb, bias, acc[0][j][i], rbase + i,      cb, Ndim);
      gemm_epi(mode, Cf, Cb, bias, acc[1][j][i], rbase + 16 + i, cb, Ndim);
    }
  }
}

// ---------------- post-GEMM elementwise/head ops ----------------
__global__ __launch_bounds__(256) void post_kernel(
    float* __restrict__ kraw, const float* __restrict__ abuf,
    float* __restrict__ vbuf, const float* __restrict__ vmix,
    const float* __restrict__ v_first,
    const float* __restrict__ k_k, const float* __restrict__ k_a,
    float* __restrict__ kkbuf) {
  long long rh = (long long)blockIdx.x * 8 + (threadIdx.x >> 5);
  if (rh >= Mrows * Hh) return;
  int lane = threadIdx.x & 31;
  int h = (int)(rh & (Hh - 1));
  long long base = rh * HDd;
  int c0 = lane, c1 = lane + 32;
  int d0 = h * HDd + c0, d1 = h * HDd + c1;

  float kr0 = kraw[base + c0], kr1 = kraw[base + c1];
  float kk0 = kr0 * k_k[d0],   kk1 = kr1 * k_k[d1];
  float ss  = wave_sum(kk0 * kk0 + kk1 * kk1);
  float rs  = rsqrtf(ss + 1e-12f);
  kkbuf[base + c0] = kk0 * rs;
  kkbuf[base + c1] = kk1 * rs;

  float a0 = abuf[base + c0], a1 = abuf[base + c1];
  kraw[base + c0] = kr0 * (1.0f + (a0 - 1.0f) * k_a[d0]);
  kraw[base + c1] = kr1 * (1.0f + (a1 - 1.0f) * k_a[d1]);

  float v0 = vbuf[base + c0], v1 = vbuf[base + c1];
  vbuf[base + c0] = v0 + vmix[base + c0] * (v_first[base + c0] - v0);
  vbuf[base + c1] = v1 + vmix[base + c1] * (v_first[base + c1] - v1);
}

// ---------------- sequential RWKV7 state scan ----------------
__global__ __launch_bounds__(64) void scan_kernel(
    const float* __restrict__ rS, const float* __restrict__ wS,
    const float* __restrict__ kS, const float* __restrict__ vS,
    const float* __restrict__ kkS, const float* __restrict__ aS,
    float* __restrict__ oS) {
  __shared__ float sr[64], sew[64], skf[64], saa[64], sbb[64];
  int bh = blockIdx.x;
  int b = bh >> 4, h = bh & 15;
  int v = threadIdx.x;
  long long headoff = (long long)h * HDd;

  float S[64];
  #pragma unroll
  for (int i = 0; i < 64; ++i) S[i] = 0.0f;

  for (int t = 0; t < Tt; ++t) {
    long long base = ((long long)b * Tt + t) * Dd + headoff;
    if (t + 8 < Tt) {
      long long nb = base + 8LL * Dd;
      __builtin_prefetch(rS + nb + v, 0, 1);
      __builtin_prefetch(wS + nb + v, 0, 1);
      __builtin_prefetch(kS + nb + v, 0, 1);
      __builtin_prefetch(vS + nb + v, 0, 1);
      __builtin_prefetch(kkS + nb + v, 0, 1);
      __builtin_prefetch(aS + nb + v, 0, 1);
    }
    sr[v]  = rS[base + v];
    sew[v] = __expf(wS[base + v]);
    skf[v] = kS[base + v];
    float kkv = kkS[base + v];
    saa[v] = -kkv;                          // a_t = -kk
    sbb[v] = kkv * aS[base + v];            // b_t = kk * a
    float vt = vS[base + v];
    __syncthreads();

    float sa = 0.0f;                        // a_t^T S (pre-decay)
    #pragma unroll
    for (int k2 = 0; k2 < 64; ++k2) sa += saa[k2] * S[k2];
    float o = 0.0f;
    #pragma unroll
    for (int k2 = 0; k2 < 64; ++k2) {
      float s = S[k2] * sew[k2] + sbb[k2] * sa + skf[k2] * vt;
      S[k2] = s;
      o += sr[k2] * s;
    }
    oS[base + v] = o;
    __syncthreads();
  }
}

// ---------------- groupnorm + residual + gate -> bf16 for W_o GEMM ----------------
__global__ __launch_bounds__(256) void gn_kernel(
    const float* __restrict__ obuf, const float* __restrict__ rbuf,
    const float* __restrict__ kbuf, const float* __restrict__ vbuf,
    const float* __restrict__ gbuf, const float* __restrict__ r_k,
    const float* __restrict__ gnw, const float* __restrict__ gnb,
    __bf16* __restrict__ ybf) {
  long long rh = (long long)blockIdx.x * 8 + (threadIdx.x >> 5);
  if (rh >= Mrows * Hh) return;
  int lane = threadIdx.x & 31;
  int h = (int)(rh & (Hh - 1));
  long long base = rh * HDd;
  int c0 = lane, c1 = lane + 32;
  int d0 = h * HDd + c0, d1 = h * HDd + c1;

  float o0 = obuf[base + c0], o1 = obuf[base + c1];
  float mean = wave_sum(o0 + o1) * (1.0f / 64.0f);
  float e0 = o0 - mean, e1 = o1 - mean;
  float var = wave_sum(e0 * e0 + e1 * e1) * (1.0f / 64.0f);
  float rstd = rsqrtf(var + 64.0f * 1e-5f);
  float on0 = e0 * rstd * gnw[d0] + gnb[d0];
  float on1 = e1 * rstd * gnw[d1] + gnb[d1];

  float dot = wave_sum(rbuf[base + c0] * kbuf[base + c0] * r_k[d0] +
                       rbuf[base + c1] * kbuf[base + c1] * r_k[d1]);
  float y0 = (on0 + dot * vbuf[base + c0]) * gbuf[base + c0];
  float y1 = (on1 + dot * vbuf[base + c1]) * gbuf[base + c1];
  ybf[base + c0] = f32_to_bf16(y0);
  ybf[base + c1] = f32_to_bf16(y1);
}

// ---------------- host ----------------
extern "C" void kernel_launch(void* const* d_in, const int* in_sizes, int n_in,
                              void* d_out, int out_size, void* d_ws, size_t ws_size,
                              hipStream_t stream) {
  const float* hs      = (const float*)d_in[0];
  const float* v_first = (const float*)d_in[1];
  const float* x_x     = (const float*)d_in[2];
  const float* k_k     = (const float*)d_in[3];
  const float* k_a     = (const float*)d_in[4];
  const float* r_k     = (const float*)d_in[5];
  const float* W[4]    = {(const float*)d_in[6], (const float*)d_in[7],
                          (const float*)d_in[8], (const float*)d_in[9]};
  const float* wla = (const float*)d_in[10]; const float* wlb = (const float*)d_in[11];
  const float* wbias = (const float*)d_in[12];
  const float* vla = (const float*)d_in[13]; const float* vlb = (const float*)d_in[14];
  const float* vbias = (const float*)d_in[15];
  const float* ala = (const float*)d_in[16]; const float* alb = (const float*)d_in[17];
  const float* abias = (const float*)d_in[18];
  const float* gla = (const float*)d_in[19]; const float* glb = (const float*)d_in[20];
  const float* gnw = (const float*)d_in[21]; const float* gnb = (const float*)d_in[22];

  char* ws = (char*)d_ws;
  size_t off = 0;
  auto alloc = [&](size_t bytes) -> void* {
    void* p = ws + off;
    off += (bytes + 255) & ~(size_t)255;
    return p;
  };

  __bf16* xs[6];
  for (int i = 0; i < 6; ++i) xs[i] = (__bf16*)alloc(MD * 2);
  __bf16* Wb[4];
  for (int i = 0; i < 4; ++i) Wb[i] = (__bf16*)alloc((size_t)Dd * Dd * 2);
  __bf16* wla_b = (__bf16*)alloc((size_t)Dd * 64 * 2);
  __bf16* wlb_b = (__bf16*)alloc((size_t)64 * Dd * 2);
  __bf16* vla_b = (__bf16*)alloc((size_t)Dd * 64 * 2);   // rank-16 zero-padded to 64
  __bf16* vlb_b = (__bf16*)alloc((size_t)64 * Dd * 2);   // rank-16 zero-padded to 64
  __bf16* ala_b = (__bf16*)alloc((size_t)Dd * 64 * 2);
  __bf16* alb_b = (__bf16*)alloc((size_t)64 * Dd * 2);
  __bf16* gla_b = (__bf16*)alloc((size_t)Dd * 128 * 2);
  __bf16* glb_b = (__bf16*)alloc((size_t)128 * Dd * 2);
  __bf16* wmid  = (__bf16*)alloc((size_t)Mrows * 64 * 2);
  __bf16* vmid  = (__bf16*)alloc((size_t)Mrows * 64 * 2);
  __bf16* amid  = (__bf16*)alloc((size_t)Mrows * 64 * 2);
  __bf16* gmid  = (__bf16*)alloc((size_t)Mrows * 128 * 2);
  __bf16* ybf   = (__bf16*)alloc(MD * 2);
  float* rbuf    = (float*)alloc(MD * 4);
  float* wbuf    = (float*)alloc(MD * 4);
  float* kbuf    = (float*)alloc(MD * 4);
  float* vbuf    = (float*)alloc(MD * 4);
  float* abuf    = (float*)alloc(MD * 4);
  float* gbuf    = (float*)alloc(MD * 4);
  float* vmixbuf = (float*)alloc(MD * 4);
  float* kkbuf   = (float*)alloc(MD * 4);
  float* obuf    = (float*)alloc(MD * 4);
  (void)ws_size; (void)in_sizes; (void)n_in; (void)out_size;

  auto cvt = [&](const float* s, __bf16* d, int n) {
    cvt_bf16x4_kernel<<<(n / 4 + 255) / 256, 256, 0, stream>>>(s, d, n / 4);
  };
  for (int i = 0; i < 4; ++i) cvt(W[i], Wb[i], Dd * Dd);
  cvt(wla, wla_b, Dd * 64);  cvt(wlb, wlb_b, 64 * Dd);
  cvt(ala, ala_b, Dd * 64);  cvt(alb, alb_b, 64 * Dd);
  cvt(gla, gla_b, Dd * 128); cvt(glb, glb_b, 128 * Dd);
  cvt_pad_kernel<<<(Dd * 64 + 255) / 256, 256, 0, stream>>>(vla, vla_b, Dd, 64, Dd, 16);
  cvt_pad_kernel<<<(64 * Dd + 255) / 256, 256, 0, stream>>>(vlb, vlb_b, 64, Dd, 16, Dd);

  prep_kernel<<<(int)((MD / 4 + 255) / 256), 256, 0, stream>>>(
      hs, x_x, xs[0], xs[1], xs[2], xs[3], xs[4], xs[5]);

  auto gemm = [&](const __bf16* A, const __bf16* Bm, float* Cf, __bf16* Cb,
                  const float* bias, int Md_, int Nd_, int Kd_, int mode) {
    if (Nd_ % 128 == 0) {
      dim3 g(Nd_ / 128, Md_ / 128);
      gemm_bf16_wmma<128><<<g, 256, 0, stream>>>(A, Bm, Cf, Cb, bias, Nd_, Kd_, mode);
    } else {
      dim3 g(Nd_ / 64, Md_ / 128);
      gemm_bf16_wmma<64><<<g, 256, 0, stream>>>(A, Bm, Cf, Cb, bias, Nd_, Kd_, mode);
    }
  };
  const int M = (int)Mrows;

  gemm(xs[0], Wb[0], rbuf, nullptr, nullptr, M, Dd, Dd, 0);      // r = xr @ W_r
  gemm(xs[2], Wb[1], kbuf, nullptr, nullptr, M, Dd, Dd, 0);      // k = xk @ W_k
  gemm(xs[3], Wb[2], vbuf, nullptr, nullptr, M, Dd, Dd, 0);      // v = xv @ W_v
  gemm(xs[1], wla_b, nullptr, wmid, nullptr, M, 64, Dd, 1);      // tanh(xw @ w_a)
  gemm(wmid, wlb_b, wbuf, nullptr, wbias, M, Dd, 64, 4);         // w decay
  gemm(xs[3], vla_b, nullptr, vmid, nullptr, M, 64, Dd, 2);      // xv @ v_a (padded)
  gemm(vmid, vlb_b, vmixbuf, nullptr, vbias, M, Dd, 64, 5);      // v_mix
  gemm(xs[4], ala_b, nullptr, amid, nullptr, M, 64, Dd, 2);      // xa @ a_a
  gemm(amid, alb_b, abuf, nullptr, abias, M, Dd, 64, 5);         // a
  gemm(xs[5], gla_b, nullptr, gmid, nullptr, M, 128, Dd, 3);     // sigmoid(xg @ g_a)
  gemm(gmid, glb_b, gbuf, nullptr, nullptr, M, Dd, 128, 0);      // g

  long long rowheads = Mrows * Hh;
  post_kernel<<<(int)((rowheads + 7) / 8), 256, 0, stream>>>(
      kbuf, abuf, vbuf, vmixbuf, v_first, k_k, k_a, kkbuf);

  scan_kernel<<<Bb * Hh, 64, 0, stream>>>(rbuf, wbuf, kbuf, vbuf, kkbuf, abuf, obuf);

  gn_kernel<<<(int)((rowheads + 7) / 8), 256, 0, stream>>>(
      obuf, rbuf, kbuf, vbuf, gbuf, r_k, gnw, gnb, ybf);

  gemm(ybf, Wb[3], (float*)d_out, nullptr, nullptr, M, Dd, Dd, 0);  // out = y @ W_o
}
